// TokenBlockCrossAttention_33294586479030
// MI455X (gfx1250) — matrix operand order repaired and checked by reading, not compiled, not run on hardware
//
#include <hip/hip_runtime.h>

#define BB 2
#define SS 2048
#define NN 1024
#define HH 1024
#define NHH 16
#define DD 64
#define SC_LD 1040   // padded LDS row stride (floats / bf16 elems) for attention
#define WT_LDS 40    // bf16 elems per weight-tile row in LDS (64B data + 16B TDM pad)

typedef __attribute__((ext_vector_type(16))) __bf16 v16bf;
typedef __attribute__((ext_vector_type(8)))  __bf16 v8bf;
typedef __attribute__((ext_vector_type(8)))  float  v8f;
typedef __attribute__((ext_vector_type(4)))  unsigned v4u;
typedef __attribute__((ext_vector_type(8)))  int    v8i;
typedef __attribute__((ext_vector_type(4)))  int    v4i;

static __device__ __forceinline__ __bf16 f2bf(float f) {
  unsigned u = __builtin_bit_cast(unsigned, f);
  unsigned r = u + 0x7FFFu + ((u >> 16) & 1u);   // round-to-nearest-even
  unsigned short h = (unsigned short)(r >> 16);
  return __builtin_bit_cast(__bf16, h);
}

static __device__ __forceinline__ v16bf cat16(v8bf lo, v8bf hi) {
  return __builtin_shufflevector(lo, hi, 0,1,2,3,4,5,6,7,8,9,10,11,12,13,14,15);
}

// ---- Tensor Data Mover: async 2D tile load Global->LDS (D# per CDNA5 ISA §8) ----
// Loads tile_dim1=64 rows x tile_dim0=32 bf16 (64B/row) from a row-major
// [1024 x 1024] bf16 matrix into LDS, with TDM padding of 16B after every
// 64B so the LDS row stride is 80B (bank-deconflicted ds_load_b128 reads).
static __device__ __forceinline__ void tdm_load_wtile(const __bf16* gsrc, unsigned lds_byte_off) {
  unsigned long long ga = (unsigned long long)(uintptr_t)gsrc;
  v4u g0;
  g0.x = 1u;                                                  // count=1 (valid), no gather
  g0.y = lds_byte_off;                                        // lds_addr [63:32]
  g0.z = (unsigned)(ga & 0xFFFFFFFFu);                        // global_addr [95:64]
  g0.w = (unsigned)((ga >> 32) & 0x01FFFFFFu) | (2u << 30);   // addr[56:32] | type=2

  v8i g1;
  g1[0] = (1 << 16)        // data_size = 1 -> 2 bytes
        | (1 << 20)        // pad_enable
        | (3 << 22)        // pad_interval = 16 DWORDs (64B)
        | (3 << 25);       // pad_amount   = 4 DWORDs (16B)
  g1[1] = (HH & 0xFFFF) << 16;                           // tensor_dim0 lo16
  g1[2] = ((HH >> 16) & 0xFFFF) | ((HH & 0xFFFF) << 16); // dim0 hi16 | dim1 lo16
  g1[3] = ((HH >> 16) & 0xFFFF) | (32 << 16);            // dim1 hi16 | tile_dim0 = 32
  g1[4] = 64;                                            // tile_dim1 = 64, tile_dim2 = 0
  g1[5] = HH;                                            // tensor_dim0_stride lo32
  g1[6] = 0;                                             // stride hi | dim1_stride lo
  g1[7] = 0;
  v4i z4 = {0, 0, 0, 0};
  v8i z8 = {0, 0, 0, 0, 0, 0, 0, 0};
  __builtin_amdgcn_tensor_load_to_lds(g0, g1, z4, z4, z8, 0);
}

// ---------------------------------------------------------------- convert ---
__global__ void cvt_f32_bf16(const float* __restrict__ in, __bf16* __restrict__ out, int n) {
  int i = blockIdx.x * blockDim.x + threadIdx.x;
  int stride = gridDim.x * blockDim.x;
  for (; i < n; i += stride) out[i] = f2bf(in[i]);
}

// ------------------------------------------------------------------- GEMM ---
// out[M x 1024] = A[M x 1024] @ W.T, W stored [1024 out][1024 in] row-major.
// Weight tiles are TDM-streamed into LDS (double-buffered, per-wave region,
// TENSORcnt-paced); WMMA B-fragments are ds_load_b128 from the padded tile.
// MODE 0: bf16 out row-major; MODE 1: bf16 out as V^T [B,NH,D,N]; MODE 2: f32 out.
template <int MODE>
__global__ void gemm_bf16(const __bf16* __restrict__ A, const __bf16* __restrict__ W,
                          void* __restrict__ out) {
  __shared__ __bf16 wtile[2][4][64 * WT_LDS];   // 2 bufs x 4 waves x 64 rows x 80B = 40 KB

  const int lane = threadIdx.x & 31;
  const int wid  = threadIdx.x >> 5;
  const int m0   = blockIdx.x * 16;
  const int ob   = blockIdx.y * 256 + wid * 64;
  const int mrow = lane & 15;
  const int kb   = (lane >> 4) * 8;    // A-fragment K sub-offset
  const int kk16 = (lane >> 4) * 8;    // B-fragment K sub-offset

  const __bf16* wbase = W + (size_t)ob * HH;   // this wave's 64 weight rows
  const unsigned lds0 = (unsigned)(uintptr_t)&wtile[0][wid][0];
  const unsigned lds1 = (unsigned)(uintptr_t)&wtile[1][wid][0];

  v8f acc[4] = {{}, {}, {}, {}};
  const __bf16* arow = A + (size_t)(m0 + mrow) * HH;

  tdm_load_wtile(wbase, lds0);                 // prime buffer 0
  int buf = 0;
  for (int k0 = 0; k0 < HH; k0 += 32, buf ^= 1) {
    if (k0 + 32 < HH) {
      tdm_load_wtile(wbase + (k0 + 32), buf ? lds0 : lds1);   // prefetch next tile
      __builtin_amdgcn_s_wait_tensorcnt(1);                   // current tile complete
    } else {
      __builtin_amdgcn_s_wait_tensorcnt(0);
    }
    __builtin_prefetch(arow + k0 + 256, 0, 0);
    v16bf afrag = cat16(*(const v8bf*)(arow + k0 + kb),
                        *(const v8bf*)(arow + k0 + kb + 16));
    const __bf16* wt = &wtile[buf][wid][0];
#pragma unroll
    for (int j = 0; j < 4; ++j) {
      const __bf16* wrow = wt + (j * 16 + mrow) * WT_LDS + kk16;
      v16bf bfrag = cat16(*(const v8bf*)wrow, *(const v8bf*)(wrow + 16));
      acc[j] = __builtin_amdgcn_wmma_f32_16x16x32_bf16(
          false, afrag, false, bfrag, (short)0, acc[j], false, false);
    }
  }

#pragma unroll
  for (int j = 0; j < 4; ++j) {
    int o = ob + j * 16 + mrow;
#pragma unroll
    for (int r = 0; r < 8; ++r) {
      int mg = m0 + r + 8 * (lane >> 4);
      float val = acc[j][r];
      if (MODE == 0) {
        ((__bf16*)out)[(size_t)mg * HH + o] = f2bf(val);
      } else if (MODE == 2) {
        ((float*)out)[(size_t)mg * HH + o] = val;
      } else {  // V transposed: [B, NH, D, N]
        int b = mg / NN, n = mg % NN;
        int h = o / DD, d = o % DD;
        ((__bf16*)out)[(size_t)((b * NHH + h) * DD + d) * NN + n] = f2bf(val);
      }
    }
  }
}

// -------------------------------------------------------------- attention ---
// One workgroup (4 waves) per (b, h, 16-row S tile).
__global__ void attn_kernel(const __bf16* __restrict__ q, const __bf16* __restrict__ k,
                            const __bf16* __restrict__ vt, const int* __restrict__ ends,
                            __bf16* __restrict__ ao) {
  extern __shared__ char smem[];
  float*  sc     = (float*)smem;                                  // [16][SC_LD] f32 scores
  __bf16* pb     = (__bf16*)(smem + 16 * SC_LD * 4);              // [16][SC_LD] bf16 probs
  int*    ends_s = (int*)(smem + 16 * SC_LD * 4 + 16 * SC_LD * 2);
  float*  red    = (float*)(ends_s + NN);                         // [16*8]
  float*  rmax   = red + 16 * 8;                                  // [16]
  float*  rsum   = rmax + 16;                                     // [16]

  const int tid  = threadIdx.x;
  const int lane = tid & 31;
  const int wid  = tid >> 5;
  const int s0   = blockIdx.x * 16;
  const int h    = blockIdx.y;
  const int b    = blockIdx.z;
  const int mrow = lane & 15;
  const int kb   = (lane >> 4) * 8;
  const int kk16 = (lane >> 4) * 8;

  for (int i = tid; i < NN; i += 128) ends_s[i] = ends[i];

  // ---- scores = (q @ k^T) / 8 : wave w covers n in [w*256, w*256+256)
  const __bf16* qrow = q + (size_t)(b * SS + s0 + mrow) * HH + h * DD;
  v16bf a0 = cat16(*(const v8bf*)(qrow + kb),       *(const v8bf*)(qrow + kb + 16));
  v16bf a1 = cat16(*(const v8bf*)(qrow + 32 + kb),  *(const v8bf*)(qrow + 32 + kb + 16));
  const __bf16* kbase = k + (size_t)b * NN * HH + h * DD;
  for (int j = 0; j < 16; ++j) {
    int n0 = wid * 256 + j * 16;
    const __bf16* krow = kbase + (size_t)(n0 + mrow) * HH;
    v16bf b0 = cat16(*(const v8bf*)(krow + kk16),      *(const v8bf*)(krow + kk16 + 16));
    v16bf b1 = cat16(*(const v8bf*)(krow + 32 + kk16), *(const v8bf*)(krow + 32 + kk16 + 16));
    v8f c = {};
    c = __builtin_amdgcn_wmma_f32_16x16x32_bf16(false, a0, false, b0, (short)0, c, false, false);
    c = __builtin_amdgcn_wmma_f32_16x16x32_bf16(false, a1, false, b1, (short)0, c, false, false);
#pragma unroll
    for (int r = 0; r < 8; ++r)
      sc[(r + 8 * (lane >> 4)) * SC_LD + n0 + mrow] = c[r] * 0.125f;
  }
  __syncthreads();

  // ---- masked softmax over n (row-parallel: 16 rows x 8 column groups)
  const int m = tid & 15;
  const int g = tid >> 4;          // 0..7, each owns 128 columns
  const int t = s0 + m;
  const float NEGF = -3.4e38f;

  float lmax = NEGF;
  for (int n = g * 128; n < g * 128 + 128; ++n) {
    float v = (ends_s[n] <= t) ? sc[m * SC_LD + n] : NEGF;
    lmax = fmaxf(lmax, v);
  }
  red[m * 8 + g] = lmax;
  __syncthreads();
  if (g == 0) {
    float mx = red[m * 8];
    for (int i = 1; i < 8; ++i) mx = fmaxf(mx, red[m * 8 + i]);
    rmax[m] = mx;
  }
  __syncthreads();
  float mx = rmax[m];
  float lsum = 0.f;
  for (int n = g * 128; n < g * 128 + 128; ++n) {
    float v = (ends_s[n] <= t) ? sc[m * SC_LD + n] : NEGF;
    lsum += __expf(v - mx);
  }
  red[m * 8 + g] = lsum;
  __syncthreads();
  if (g == 0) {
    float s = 0.f;
    for (int i = 0; i < 8; ++i) s += red[m * 8 + i];
    rsum[m] = s;
  }
  __syncthreads();
  float inv = 1.0f / rsum[m];
  for (int n = g * 128; n < g * 128 + 128; ++n) {
    float v = (ends_s[n] <= t) ? sc[m * SC_LD + n] : NEGF;
    pb[m * SC_LD + n] = f2bf(__expf(v - mx) * inv);
  }
  __syncthreads();

  // ---- O = P @ V : wave w owns d-tile [w*16, w*16+16); K dim = n (1024)
  v8f oacc = {};
  const __bf16* vrow = vt + (size_t)((b * NHH + h) * DD + wid * 16 + mrow) * NN;
  for (int n0 = 0; n0 < NN; n0 += 32) {
    int nb = n0 + kb;
    v16bf pa = cat16(*(const v8bf*)(pb + mrow * SC_LD + nb),
                     *(const v8bf*)(pb + mrow * SC_LD + nb + 16));
    v16bf vb = cat16(*(const v8bf*)(vrow + n0 + kk16),
                     *(const v8bf*)(vrow + n0 + kk16 + 16));
    oacc = __builtin_amdgcn_wmma_f32_16x16x32_bf16(false, pa, false, vb, (short)0, oacc, false, false);
  }
#pragma unroll
  for (int r = 0; r < 8; ++r) {
    int srow = s0 + r + 8 * (lane >> 4);
    ao[(size_t)(b * SS + srow) * HH + h * DD + wid * 16 + mrow] = f2bf(oacc[r]);
  }
}

// ----------------------------------------------------------------- launch ---
extern "C" void kernel_launch(void* const* d_in, const int* in_sizes, int n_in,
                              void* d_out, int out_size, void* d_ws, size_t ws_size,
                              hipStream_t stream) {
  const float* token_q    = (const float*)d_in[0];
  const float* block_kv   = (const float*)d_in[1];
  const int*   block_ends = (const int*)d_in[2];
  const float* Wq = (const float*)d_in[3];
  const float* Wk = (const float*)d_in[4];
  const float* Wv = (const float*)d_in[5];
  const float* Wo = (const float*)d_in[6];

  char* ws = (char*)d_ws;
  size_t off = 0;
  auto alloc = [&](size_t nelem) { __bf16* p = (__bf16*)(ws + off); off += nelem * 2; return p; };

  __bf16* tq_bf = alloc((size_t)BB * SS * HH);
  __bf16* kv_bf = alloc((size_t)BB * NN * HH);
  __bf16* wq_bf = alloc((size_t)HH * HH);
  __bf16* wk_bf = alloc((size_t)HH * HH);
  __bf16* wv_bf = alloc((size_t)HH * HH);
  __bf16* wo_bf = alloc((size_t)HH * HH);
  __bf16* q_bf  = alloc((size_t)BB * SS * HH);   // [B,S,H]
  __bf16* k_bf  = alloc((size_t)BB * NN * HH);   // [B,N,H]
  __bf16* vt_bf = alloc((size_t)BB * NN * HH);   // [B,NH,D,N]
  __bf16* ao_bf = alloc((size_t)BB * SS * HH);   // [B,S,H]

  cvt_f32_bf16<<<1024, 256, 0, stream>>>(token_q,  tq_bf, BB * SS * HH);
  cvt_f32_bf16<<<1024, 256, 0, stream>>>(block_kv, kv_bf, BB * NN * HH);
  cvt_f32_bf16<<<512,  256, 0, stream>>>(Wq, wq_bf, HH * HH);
  cvt_f32_bf16<<<512,  256, 0, stream>>>(Wk, wk_bf, HH * HH);
  cvt_f32_bf16<<<512,  256, 0, stream>>>(Wv, wv_bf, HH * HH);
  cvt_f32_bf16<<<512,  256, 0, stream>>>(Wo, wo_bf, HH * HH);

  gemm_bf16<0><<<dim3(BB * SS / 16, 4), 128, 0, stream>>>(tq_bf, wq_bf, q_bf);
  gemm_bf16<0><<<dim3(BB * NN / 16, 4), 128, 0, stream>>>(kv_bf, wk_bf, k_bf);
  gemm_bf16<1><<<dim3(BB * NN / 16, 4), 128, 0, stream>>>(kv_bf, wv_bf, vt_bf);

  size_t smem = (size_t)16 * SC_LD * 4 + (size_t)16 * SC_LD * 2 + NN * 4 + (16 * 8 + 32) * 4;
  attn_kernel<<<dim3(SS / 16, NHH, BB), 128, smem, stream>>>(q_bf, k_bf, vt_bf, block_ends, ao_bf);

  gemm_bf16<2><<<dim3(BB * SS / 16, 4), 128, 0, stream>>>(ao_bf, wo_bf, d_out);
}
